// Block_22917945491932
// MI455X (gfx1250) — matrix operand order repaired and checked by reading, not compiled, not run on hardware
//
#include <hip/hip_runtime.h>
#include <hip/hip_bf16.h>
#include <stdint.h>

#define EMB   1024
#define HEADS 16
#define HDIM  64
#define BATCH 2
#define SEQ   2048
#define MROWS (BATCH * SEQ)   // 4096
#define QKVN  (3 * EMB)       // 3072
#define FFN   (4 * EMB)       // 4096
#define LNCH  16              // layernorm T-chunks

typedef __bf16 bf16_t;
typedef __attribute__((ext_vector_type(16))) __bf16    v16bf;
typedef __attribute__((ext_vector_type(8)))  float     v8f;
typedef __attribute__((ext_vector_type(4)))  uint32_t  u32x4;   // 16-byte copy unit
typedef __attribute__((ext_vector_type(4)))  int       v4i;     // builtin's pointee

// CDNA5 async global->LDS path (ASYNCcnt-tracked), with safe fallback.
#if defined(__gfx1250__) && __has_builtin(__builtin_amdgcn_global_load_async_to_lds_b128)
#define USE_ASYNC_LDS 1
#else
#define USE_ASYNC_LDS 0
#endif

union FragA {
  v16bf    v;
  uint16_t u[16];
  u32x4    q[2];
};

__device__ __forceinline__ uint16_t f2bf(float f) {
  uint32_t u = __builtin_bit_cast(uint32_t, f);
  uint32_t r = u + 0x7FFFu + ((u >> 16) & 1u);   // round-to-nearest-even
  return (uint16_t)(r >> 16);
}

__device__ __forceinline__ v8f wmma_bf16(v16bf a, v16bf b, v8f c) {
  return __builtin_amdgcn_wmma_f32_16x16x32_bf16(false, a, false, b, (short)0, c,
                                                 false, false);
}

#if USE_ASYNC_LDS
__device__ __forceinline__ void g2l_async16(const uint16_t* g, uint16_t* l) {
  // Diagnostic-verified param 0 type: 'int4 __device__ *' (addrspace(1));
  // param 1 is the LDS-side mirror (addrspace(3)).
  __builtin_amdgcn_global_load_async_to_lds_b128(
      (__attribute__((address_space(1))) v4i*)g,
      (__attribute__((address_space(3))) v4i*)l, 0, 0);
}
__device__ __forceinline__ void wait_async0() {
  asm volatile("s_wait_asynccnt 0x0" ::: "memory");
}
#endif

// ---------------------------------------------------------------------------
// LayerNorm over the SEQUENCE axis, unbiased variance. Two-stage reduction:
// 16 T-chunks per (b,c) -> 1024 waves of partial sums, then tiny finish pass.
// ---------------------------------------------------------------------------
__global__ void ln_part_kernel(const float* __restrict__ x,
                               float* __restrict__ ps, float* __restrict__ ps2) {
  int idx = blockIdx.x * blockDim.x + threadIdx.x;      // over BATCH*EMB*LNCH
  if (idx >= BATCH * EMB * LNCH) return;
  int bc    = idx % (BATCH * EMB);
  int chunk = idx / (BATCH * EMB);
  int b = bc / EMB, c = bc % EMB;
  const float* p = x + (size_t)b * SEQ * EMB + (size_t)chunk * (SEQ / LNCH) * EMB + c;
  float s = 0.f, s2 = 0.f;
  for (int t = 0; t < SEQ / LNCH; ++t) {
    float v = p[(size_t)t * EMB];
    s += v; s2 += v * v;
  }
  ps[idx] = s; ps2[idx] = s2;
}

__global__ void ln_finish_kernel(const float* __restrict__ ps,
                                 const float* __restrict__ ps2,
                                 float* __restrict__ mean,
                                 float* __restrict__ rstd) {
  int idx = blockIdx.x * blockDim.x + threadIdx.x;      // over BATCH*EMB
  if (idx >= BATCH * EMB) return;
  float s = 0.f, s2 = 0.f;
  #pragma unroll
  for (int k = 0; k < LNCH; ++k) {
    s  += ps [(size_t)k * BATCH * EMB + idx];
    s2 += ps2[(size_t)k * BATCH * EMB + idx];
  }
  float m   = s * (1.0f / SEQ);
  float var = (s2 - s * m) * (1.0f / (SEQ - 1));        // ddof=1
  mean[idx] = m;
  rstd[idx] = rsqrtf(var + 1e-5f);
}

__global__ void ln_apply_kernel(const float* __restrict__ x,
                                const float* __restrict__ mean,
                                const float* __restrict__ rstd,
                                const float* __restrict__ g,
                                const float* __restrict__ be,
                                uint16_t* __restrict__ out) {
  size_t i = (size_t)blockIdx.x * blockDim.x + threadIdx.x;
  if (i >= (size_t)MROWS * EMB) return;
  int c  = (int)(i % EMB);
  int bt = (int)(i / EMB);
  int b  = bt / SEQ;
  int sc = b * EMB + c;
  float v = (x[i] - mean[sc]) * rstd[sc] * g[c] + be[c];
  out[i] = f2bf(v);
}

// ---------------------------------------------------------------------------
// Weight packing: fp32 -> bf16, transposed to [N][K].
// ---------------------------------------------------------------------------
__global__ void pack_qkvT_kernel(const float* __restrict__ Wq,
                                 const float* __restrict__ Wk,
                                 const float* __restrict__ Wv,
                                 uint16_t* __restrict__ WqkvT) {  // [3072][1024]
  int i = blockIdx.x * blockDim.x + threadIdx.x;
  if (i >= QKVN * EMB) return;
  int c = i % EMB;
  int j = i / EMB;
  int kind = j / EMB;
  int w = j % EMB;
  int h = w / HDIM, d = w % HDIM;
  const float* W = (kind == 0) ? Wq : (kind == 1) ? Wk : Wv;
  WqkvT[(size_t)j * EMB + c] = f2bf(W[((size_t)h * EMB + c) * HDIM + d]);
}

__global__ void packT_kernel(const float* __restrict__ W,  // [K][N] f32
                             uint16_t* __restrict__ Wt,    // [N][K] bf16
                             int K, int N) {
  size_t i = (size_t)blockIdx.x * blockDim.x + threadIdx.x;
  if (i >= (size_t)K * N) return;
  int k = (int)(i / N), n = (int)(i % N);
  Wt[(size_t)n * K + k] = f2bf(W[i]);
}

// ---------------------------------------------------------------------------
// V transpose: qkv[b*T+t][2048 + h*64 + d]  ->  Vt[(b*H+h)*64 + d][SEQ] (bf16)
// LDS-tiled 64x64 so both global directions are 128B-coalesced b128 ops.
// ---------------------------------------------------------------------------
__global__ __launch_bounds__(256)
void packVt_kernel(const uint16_t* __restrict__ qkv, uint16_t* __restrict__ Vt) {
  __shared__ __align__(16) uint16_t S[64][72];    // 72-stride: dodge bank conflicts

  const int tid   = threadIdx.x;
  const int tBase = blockIdx.x * 64;              // SEQ/64 tiles
  const int h     = blockIdx.y;
  const int b     = blockIdx.z;

  // Load 64(t) x 64(d) tile: 4 threads/row, 16 ushort each.
  {
    const int tl = tid >> 2, c0 = (tid & 3) * 16;
    const uint16_t* src =
        qkv + (size_t)(b * SEQ + tBase + tl) * QKVN + 2 * EMB + h * HDIM + c0;
    *(u32x4*)&S[tl][c0]     = *(const u32x4*)(src);
    *(u32x4*)&S[tl][c0 + 8] = *(const u32x4*)(src + 8);
  }
  __syncthreads();

  // Store transposed: 4 threads/d-row, 16 t-values each (gathered from LDS).
  {
    const int dl = tid >> 2, t0 = (tid & 3) * 16;
    FragA tmp;
    #pragma unroll
    for (int e = 0; e < 16; ++e) tmp.u[e] = S[t0 + e][dl];
    uint16_t* dst =
        Vt + ((size_t)(b * HEADS + h) * HDIM + dl) * SEQ + tBase + t0;
    *(u32x4*)(dst)     = tmp.q[0];
    *(u32x4*)(dst + 8) = tmp.q[1];
  }
}

// ---------------------------------------------------------------------------
// Tiled bf16 GEMM:  C[M,N] = A[M,K] @ Bt[N,K]^T   (f32 accumulate)
// Block = 256 threads = 8 waves; tile 128(M) x 128(N), BK = 32, double-
// buffered LDS. Wave (wm,wn) owns 32x64 -> 8 v_wmma per K-step with only
// 12 ds_load_b128. Async global->LDS when available (ASYNCcnt), else a
// register-staged software pipeline. Epilogue fuses bias/relu/residual.
// ---------------------------------------------------------------------------
__global__ __launch_bounds__(256)
void gemm_bf16_kernel(const uint16_t* __restrict__ A,
                      const uint16_t* __restrict__ Bt,
                      int M, int N, int K,
                      const float* __restrict__ bias, int relu,
                      const float* __restrict__ residual,
                      float* __restrict__ outF, uint16_t* __restrict__ outH) {
  __shared__ __align__(16) uint16_t As[2][128][32];   // 16 KB
  __shared__ __align__(16) uint16_t Bs[2][128][32];   // 16 KB

  const int tid  = threadIdx.x;
  const int lane = tid & 31;
  const int w    = tid >> 5;
  const int wm   = w & 3;           // 32-row sub-tile
  const int wn   = w >> 2;          // 64-col sub-tile
  const int rowBase = blockIdx.y * 128;
  const int colBase = blockIdx.x * 128;

  v8f acc[2][4];
  #pragma unroll
  for (int g = 0; g < 2; ++g)
    #pragma unroll
    for (int t = 0; t < 4; ++t) acc[g][t] = (v8f){0,0,0,0,0,0,0,0};

  // Cooperative copy mapping: 128 rows x 32 cols, 16 bf16 (2 x b128) / thread.
  const int cr = tid >> 1;
  const int cc = (tid & 1) * 16;
  const uint16_t* gA = A  + (size_t)(rowBase + cr) * K + cc;
  const uint16_t* gB = Bt + (size_t)(colBase + cr) * K + cc;

  const int mrow0 = wm * 32 + (lane & 15);
  const int kbA   = (lane < 16) ? 0 : 16;
  const int kbB   = (lane < 16) ? 0 : 16;

  auto compute = [&](int buf) {
    FragA fa0, fa1;
    fa0.q[0] = *(const u32x4*)&As[buf][mrow0][kbA];
    fa0.q[1] = *(const u32x4*)&As[buf][mrow0][kbA + 8];
    fa1.q[0] = *(const u32x4*)&As[buf][mrow0 + 16][kbA];
    fa1.q[1] = *(const u32x4*)&As[buf][mrow0 + 16][kbA + 8];
    #pragma unroll
    for (int t = 0; t < 4; ++t) {
      FragA fb;
      const int n = wn * 64 + t * 16 + (lane & 15);
      fb.q[0] = *(const u32x4*)&Bs[buf][n][kbB];
      fb.q[1] = *(const u32x4*)&Bs[buf][n][kbB + 8];
      acc[0][t] = wmma_bf16(fa0.v, fb.v, acc[0][t]);
      acc[1][t] = wmma_bf16(fa1.v, fb.v, acc[1][t]);
    }
  };

  const int KT = K >> 5;

#if USE_ASYNC_LDS
  auto loadTile = [&](int k0, int buf) {
    g2l_async16(gA + k0,     &As[buf][cr][cc]);
    g2l_async16(gA + k0 + 8, &As[buf][cr][cc + 8]);
    g2l_async16(gB + k0,     &Bs[buf][cr][cc]);
    g2l_async16(gB + k0 + 8, &Bs[buf][cr][cc + 8]);
  };
  loadTile(0, 0);
  wait_async0();
  __syncthreads();
  for (int kt = 0; kt < KT; ++kt) {
    const int buf = kt & 1;
    if (kt + 1 < KT) loadTile((kt + 1) << 5, buf ^ 1);   // overlaps compute
    compute(buf);
    wait_async0();
    __syncthreads();
  }
#else
  u32x4 rA0, rA1, rB0, rB1;
  auto gload = [&](int k0) {
    rA0 = *(const u32x4*)(gA + k0);
    rA1 = *(const u32x4*)(gA + k0 + 8);
    rB0 = *(const u32x4*)(gB + k0);
    rB1 = *(const u32x4*)(gB + k0 + 8);
  };
  auto sstore = [&](int buf) {
    *(u32x4*)&As[buf][cr][cc]     = rA0;
    *(u32x4*)&As[buf][cr][cc + 8] = rA1;
    *(u32x4*)&Bs[buf][cr][cc]     = rB0;
    *(u32x4*)&Bs[buf][cr][cc + 8] = rB1;
  };
  gload(0); sstore(0);
  __syncthreads();
  for (int kt = 0; kt < KT; ++kt) {
    const int buf = kt & 1;
    if (kt + 1 < KT) gload((kt + 1) << 5);               // overlaps compute
    compute(buf);
    if (kt + 1 < KT) sstore(buf ^ 1);
    __syncthreads();
  }
#endif

  // Epilogue: lane holds column n, rows r + (lane>=16 ? 8 : 0).
  const int n    = lane & 15;
  const int rOff = (lane < 16) ? 0 : 8;
  #pragma unroll
  for (int g = 0; g < 2; ++g) {
    #pragma unroll
    for (int t = 0; t < 4; ++t) {
      const int col = colBase + wn * 64 + t * 16 + n;
      const float bv = bias ? bias[col] : 0.0f;
      #pragma unroll
      for (int r = 0; r < 8; ++r) {
        const int row = rowBase + wm * 32 + g * 16 + r + rOff;
        float v = acc[g][t][r] + bv;
        if (relu) v = fmaxf(v, 0.0f);
        const size_t gi = (size_t)row * N + col;
        if (residual) v += residual[gi];
        if (outF) outF[gi] = v;
        if (outH) outH[gi] = f2bf(v);
      }
    }
  }
}

// ---------------------------------------------------------------------------
// Flash attention: one wave per (b, h, 16-query tile); 32-key inner tiles.
// S = Q.Kt via 4 WMMA (Kt frags contiguous along d in qkv); online softmax
// with 16-lane shfl_xor row reductions; P bounced via LDS to A-layout;
// P@V via 4 WMMA with contiguous b128 frags from transposed Vt.
// ---------------------------------------------------------------------------
__global__ __launch_bounds__(32)
void attn_kernel(const uint16_t* __restrict__ qkv,
                 const uint16_t* __restrict__ Vt,
                 const float* __restrict__ x,
                 float* __restrict__ x1) {
  __shared__ __align__(16) uint16_t P[16][32];

  const int lane  = threadIdx.x;
  const int qBase = blockIdx.x * 16;
  const int h     = blockIdx.y;
  const int b     = blockIdx.z;
  const size_t rs = QKVN;

  const int m    = lane & 15;
  const int kbA  = (lane < 16) ? 0 : 16;
  const int kbB  = (lane < 16) ? 0 : 16;
  const int n    = lane & 15;
  const int rOff = (lane < 16) ? 0 : 8;

  const uint16_t* qrow = qkv + (size_t)(b * SEQ + qBase + m) * rs + h * HDIM;
  FragA aq0, aq1;
  aq0.q[0] = *(const u32x4*)(qrow + kbA);
  aq0.q[1] = *(const u32x4*)(qrow + kbA + 8);
  aq1.q[0] = *(const u32x4*)(qrow + 32 + kbA);
  aq1.q[1] = *(const u32x4*)(qrow + 32 + kbA + 8);

  const uint16_t* vbase = Vt + (size_t)(b * HEADS + h) * HDIM * SEQ;

  v8f o[4];
  #pragma unroll
  for (int t = 0; t < 4; ++t) o[t] = (v8f){0,0,0,0,0,0,0,0};
  float mrun[8], lrun[8];
  #pragma unroll
  for (int r = 0; r < 8; ++r) { mrun[r] = -3e38f; lrun[r] = 0.0f; }

  const float scale = 0.03125f;   // EMB^-0.5 (reference scales by C=1024)

  for (int sBase = 0; sBase <= qBase + 15; sBase += 32) {
    v8f S0 = {0,0,0,0,0,0,0,0};   // keys sBase .. sBase+15
    v8f S1 = {0,0,0,0,0,0,0,0};   // keys sBase+16 .. sBase+31

    #pragma unroll
    for (int kh = 0; kh < 2; ++kh) {
      const FragA& aq = kh ? aq1 : aq0;
      const uint16_t* k0p =
          qkv + (size_t)(b * SEQ + sBase + n) * rs + EMB + h * HDIM + kh * 32 + kbB;
      FragA bk0, bk1;
      bk0.q[0] = *(const u32x4*)(k0p);
      bk0.q[1] = *(const u32x4*)(k0p + 8);
      const uint16_t* k1p = k0p + 16 * rs;
      bk1.q[0] = *(const u32x4*)(k1p);
      bk1.q[1] = *(const u32x4*)(k1p + 8);
      S0 = wmma_bf16(aq.v, bk0.v, S0);
      S1 = wmma_bf16(aq.v, bk1.v, S1);
    }

    #pragma unroll
    for (int r = 0; r < 8; ++r) {
      const int q = qBase + r + rOff;
      float s0 = S0[r] * scale; if (sBase + n      > q) s0 = -3e38f;
      float s1 = S1[r] * scale; if (sBase + 16 + n > q) s1 = -3e38f;
      float lm = fmaxf(s0, s1);
      #pragma unroll
      for (int off = 1; off < 16; off <<= 1)
        lm = fmaxf(lm, __shfl_xor(lm, off, 32));
      const float mnew  = fmaxf(mrun[r], lm);
      const float alpha = __expf(mrun[r] - mnew);
      const float p0 = __expf(s0 - mnew);
      const float p1 = __expf(s1 - mnew);
      float rsum = p0 + p1;
      #pragma unroll
      for (int off = 1; off < 16; off <<= 1)
        rsum += __shfl_xor(rsum, off, 32);
      lrun[r] = lrun[r] * alpha + rsum;
      mrun[r] = mnew;
      #pragma unroll
      for (int t = 0; t < 4; ++t) o[t][r] *= alpha;
      P[r + rOff][n]      = f2bf(p0);
      P[r + rOff][16 + n] = f2bf(p1);
    }
    __syncthreads();

    FragA ap;
    ap.q[0] = *(const u32x4*)&P[m][kbA];
    ap.q[1] = *(const u32x4*)&P[m][kbA + 8];

    #pragma unroll
    for (int t = 0; t < 4; ++t) {
      FragA fv;
      const uint16_t* vp = vbase + (size_t)(t * 16 + n) * SEQ + sBase + kbB;
      fv.q[0] = *(const u32x4*)(vp);
      fv.q[1] = *(const u32x4*)(vp + 8);
      o[t] = wmma_bf16(ap.v, fv.v, o[t]);
    }
    __syncthreads();
  }

  #pragma unroll
  for (int r = 0; r < 8; ++r) {
    const float inv = 1.0f / lrun[r];
    const int row = b * SEQ + qBase + r + rOff;
    #pragma unroll
    for (int t = 0; t < 4; ++t) {
      const size_t gi = (size_t)row * EMB + h * HDIM + t * 16 + n;
      x1[gi] = x[gi] + o[t][r] * inv;
    }
  }
}

// ---------------------------------------------------------------------------
extern "C" void kernel_launch(void* const* d_in, const int* in_sizes, int n_in,
                              void* d_out, int out_size, void* d_ws, size_t ws_size,
                              hipStream_t stream) {
  const float* x   = (const float*)d_in[0];
  const float* Wq  = (const float*)d_in[1];
  const float* Wk  = (const float*)d_in[2];
  const float* Wv  = (const float*)d_in[3];
  const float* W1  = (const float*)d_in[4];
  const float* b1  = (const float*)d_in[5];
  const float* W2  = (const float*)d_in[6];
  const float* b2  = (const float*)d_in[7];
  const float* g1  = (const float*)d_in[8];
  const float* be1 = (const float*)d_in[9];
  const float* g2  = (const float*)d_in[10];
  const float* be2 = (const float*)d_in[11];
  float* out = (float*)d_out;

  char* ws = (char*)d_ws;
  size_t off = 0;
  auto carve = [&](size_t bytes) -> void* {
    void* p = ws + off;
    off += (bytes + 255) & ~(size_t)255;
    return p;
  };

  float*    mean1 = (float*)carve((size_t)BATCH * EMB * 4);
  float*    rstd1 = (float*)carve((size_t)BATCH * EMB * 4);
  float*    mean2 = (float*)carve((size_t)BATCH * EMB * 4);
  float*    rstd2 = (float*)carve((size_t)BATCH * EMB * 4);
  float*    ps    = (float*)carve((size_t)BATCH * EMB * LNCH * 4);
  float*    ps2   = (float*)carve((size_t)BATCH * EMB * LNCH * 4);
  uint16_t* ln1h  = (uint16_t*)carve((size_t)MROWS * EMB * 2);
  uint16_t* ln2h  = (uint16_t*)carve((size_t)MROWS * EMB * 2);
  uint16_t* WqkvT = (uint16_t*)carve((size_t)QKVN * EMB * 2);
  uint16_t* W1T   = (uint16_t*)carve((size_t)FFN * EMB * 2);
  uint16_t* W2T   = (uint16_t*)carve((size_t)EMB * FFN * 2);
  uint16_t* qkvh  = (uint16_t*)carve((size_t)MROWS * QKVN * 2);
  uint16_t* Vth   = (uint16_t*)carve((size_t)BATCH * EMB * SEQ * 2);
  uint16_t* h1h   = (uint16_t*)carve((size_t)MROWS * FFN * 2);
  float*    x1    = (float*)carve((size_t)MROWS * EMB * 4);
  (void)in_sizes; (void)n_in; (void)out_size; (void)ws_size;

  // Weight packing (bf16, [N][K])
  pack_qkvT_kernel<<<(QKVN * EMB + 255) / 256, 256, 0, stream>>>(Wq, Wk, Wv, WqkvT);
  packT_kernel<<<(unsigned)(((size_t)EMB * FFN + 255) / 256), 256, 0, stream>>>(W1, W1T, EMB, FFN);
  packT_kernel<<<(unsigned)(((size_t)FFN * EMB + 255) / 256), 256, 0, stream>>>(W2, W2T, FFN, EMB);

  // ln1(x)
  ln_part_kernel<<<(BATCH * EMB * LNCH + 255) / 256, 256, 0, stream>>>(x, ps, ps2);
  ln_finish_kernel<<<(BATCH * EMB + 255) / 256, 256, 0, stream>>>(ps, ps2, mean1, rstd1);
  ln_apply_kernel<<<(unsigned)(((size_t)MROWS * EMB + 255) / 256), 256, 0, stream>>>(
      x, mean1, rstd1, g1, be1, ln1h);

  // qkv = ln1 @ Wqkv   -> bf16 [MROWS][3072]
  gemm_bf16_kernel<<<dim3(QKVN / 128, MROWS / 128), 256, 0, stream>>>(
      ln1h, WqkvT, MROWS, QKVN, EMB, nullptr, 0, nullptr, nullptr, qkvh);

  // Vt transpose for contiguous PV fragments
  packVt_kernel<<<dim3(SEQ / 64, HEADS, BATCH), 256, 0, stream>>>(qkvh, Vth);

  // x1 = x + attention(qkv)
  attn_kernel<<<dim3(SEQ / 16, HEADS, BATCH), 32, 0, stream>>>(qkvh, Vth, x, x1);

  // ln2(x1)
  ln_part_kernel<<<(BATCH * EMB * LNCH + 255) / 256, 256, 0, stream>>>(x1, ps, ps2);
  ln_finish_kernel<<<(BATCH * EMB + 255) / 256, 256, 0, stream>>>(ps, ps2, mean2, rstd2);
  ln_apply_kernel<<<(unsigned)(((size_t)MROWS * EMB + 255) / 256), 256, 0, stream>>>(
      x1, mean2, rstd2, g2, be2, ln2h);

  // h1 = relu(ln2 @ W1 + b1)  -> bf16 [MROWS][4096]
  gemm_bf16_kernel<<<dim3(FFN / 128, MROWS / 128), 256, 0, stream>>>(
      ln2h, W1T, MROWS, FFN, EMB, b1, 1, nullptr, nullptr, h1h);

  // out = x1 + (h1 @ W2 + b2)
  gemm_bf16_kernel<<<dim3(EMB / 128, MROWS / 128), 256, 0, stream>>>(
      h1h, W2T, MROWS, EMB, FFN, b2, 0, x1, out, nullptr);
}
